// Model_78451872629013
// MI455X (gfx1250) — matrix operand order repaired and checked by reading, not compiled, not run on hardware
//
#include <hip/hip_runtime.h>
#include <hip/hip_bf16.h>
#include <math.h>

// ---------------------------------------------------------------------------
// PCT-style point transformer for MI455X (gfx1250).
// Activations are [B*N, C] row-major (r = b*2048+n). All matmuls use WMMA
// f16-in / f32-acc GEMMs (v_wmma_f32_16x16x32_f16), 128x128x32 tiles,
// 256 threads = 8 waves (2x4), 8 wmma per wave per K-step.
// Regular shapes take a software-pipelined kernel (register-staged global
// prefetch of tile k+1 overlapping the wmma of tile k + L2 prefetch of k+2);
// ragged shapes (K=3, K=16, M=8, O<128) take the guarded general kernel.
// ---------------------------------------------------------------------------

typedef _Float16 f16;
typedef __attribute__((ext_vector_type(16))) _Float16 v16h;
typedef __attribute__((ext_vector_type(4)))  _Float16 v4h;
typedef __attribute__((ext_vector_type(8)))  float    v8f;

#define MTOT 16384   // B*N
#define NPTS 2048
#define NBATCH 8
#define LDSLD 40     // LDS row stride in halves: 80B rows -> every frag chunk 16B aligned

// ---------------------------------------------------------------------------
// Shared fragment->wmma helper (device inline)
// A-frag layout (ISA 7.12.2): lane m=lane&15, kh=lane>>4;
//   elem j<8 -> K=kh*8+j ; j>=8 -> K=16+kh*8+(j-8)
// B-frag: same K order as A-frag
// ---------------------------------------------------------------------------
__device__ __forceinline__ v16h frag_a(const f16* As, int row, int kh) {
    const v4h* ar = (const v4h*)(&As[row * LDSLD]);
    v4h a0 = ar[kh * 2 + 0];
    v4h a1 = ar[kh * 2 + 1];
    v4h a2 = ar[4 + kh * 2 + 0];
    v4h a3 = ar[4 + kh * 2 + 1];
    v16h a;
    #pragma unroll
    for (int j = 0; j < 4; ++j) {
        a[j] = a0[j]; a[4 + j] = a1[j]; a[8 + j] = a2[j]; a[12 + j] = a3[j];
    }
    return a;
}
__device__ __forceinline__ v16h frag_b(const f16* Bs, int row, int kh) {
    const v4h* br = (const v4h*)(&Bs[row * LDSLD]);
    v4h b0 = br[kh * 2 + 0];
    v4h b1 = br[kh * 2 + 1];
    v4h b2 = br[4 + kh * 2 + 0];
    v4h b3 = br[4 + kh * 2 + 1];
    v16h b;
    #pragma unroll
    for (int j = 0; j < 4; ++j) {
        b[j] = b0[j]; b[4 + j] = b1[j]; b[8 + j] = b2[j]; b[12 + j] = b3[j];
    }
    return b;
}

// ---------------------------------------------------------------------------
// Pipelined GEMM for fully-regular shapes: M%128==0, O%128==0, K%32==0,
// lda%4==0, ldb%4==0. No bounds checks anywhere.
//   TA=0: A is [M,K].  TA=1: A is [K,M].  TB=0: B is [O,K].  TB=1: B is [K,O].
// ---------------------------------------------------------------------------
template<int TA, int TB>
__global__ __launch_bounds__(256)
void gemm_wmma_pipe(const float* __restrict__ A, const float* __restrict__ Bm,
                    const float* __restrict__ bias, float* __restrict__ C,
                    int M, int K, int O, int lda, int ldb, int ldc)
{
    __shared__ f16 As[128 * LDSLD];   // As[m][k]
    __shared__ f16 Bs[128 * LDSLD];   // Bs[o][k]

    const int tid    = threadIdx.x;
    const int wave   = tid >> 5;
    const int lane   = tid & 31;
    const int lane15 = lane & 15;
    const int kh     = lane >> 4;
    const int wr     = wave >> 2;     // 0..1
    const int wc     = wave & 3;      // 0..3
    const int m0     = blockIdx.y * 128;
    const int o0     = blockIdx.x * 128;

    v8f acc[4][2];
    #pragma unroll
    for (int f = 0; f < 4; ++f)
        #pragma unroll
        for (int g = 0; g < 2; ++g)
            #pragma unroll
            for (int j = 0; j < 8; ++j) acc[f][g][j] = 0.0f;

    float4 rA[4], rB[4];

    // ---- register-stage loaders (no guards: shape is regular) ----
    auto loadA = [&](int k0) {
        #pragma unroll
        for (int it = 0; it < 4; ++it) {
            int c = tid + 256 * it;
            if (TA == 0) {
                int m = c >> 3, k4 = c & 7;
                rA[it] = *(const float4*)(A + (size_t)(m0 + m) * lda + (k0 + k4 * 4));
            } else {
                int k = c >> 5, m4 = c & 31;
                rA[it] = *(const float4*)(A + (size_t)(k0 + k) * lda + (m0 + m4 * 4));
            }
        }
    };
    auto loadB = [&](int k0) {
        #pragma unroll
        for (int it = 0; it < 4; ++it) {
            int c = tid + 256 * it;
            if (TB == 0) {
                int o = c >> 3, k4 = c & 7;
                rB[it] = *(const float4*)(Bm + (size_t)(o0 + o) * ldb + (k0 + k4 * 4));
            } else {
                int k = c >> 5, o4 = c & 31;
                rB[it] = *(const float4*)(Bm + (size_t)(k0 + k) * ldb + (o0 + o4 * 4));
            }
        }
    };
    auto storeA = [&]() {
        #pragma unroll
        for (int it = 0; it < 4; ++it) {
            int c = tid + 256 * it;
            if (TA == 0) {
                int m = c >> 3, k4 = c & 7;
                v4h h; h[0] = (f16)rA[it].x; h[1] = (f16)rA[it].y;
                       h[2] = (f16)rA[it].z; h[3] = (f16)rA[it].w;
                *(v4h*)(&As[m * LDSLD + k4 * 4]) = h;
            } else {
                int k = c >> 5, m4 = c & 31;
                As[(m4 * 4 + 0) * LDSLD + k] = (f16)rA[it].x;
                As[(m4 * 4 + 1) * LDSLD + k] = (f16)rA[it].y;
                As[(m4 * 4 + 2) * LDSLD + k] = (f16)rA[it].z;
                As[(m4 * 4 + 3) * LDSLD + k] = (f16)rA[it].w;
            }
        }
    };
    auto storeB = [&]() {
        #pragma unroll
        for (int it = 0; it < 4; ++it) {
            int c = tid + 256 * it;
            if (TB == 0) {
                int o = c >> 3, k4 = c & 7;
                v4h h; h[0] = (f16)rB[it].x; h[1] = (f16)rB[it].y;
                       h[2] = (f16)rB[it].z; h[3] = (f16)rB[it].w;
                *(v4h*)(&Bs[o * LDSLD + k4 * 4]) = h;
            } else {
                int k = c >> 5, o4 = c & 31;
                Bs[(o4 * 4 + 0) * LDSLD + k] = (f16)rB[it].x;
                Bs[(o4 * 4 + 1) * LDSLD + k] = (f16)rB[it].y;
                Bs[(o4 * 4 + 2) * LDSLD + k] = (f16)rB[it].z;
                Bs[(o4 * 4 + 3) * LDSLD + k] = (f16)rB[it].w;
            }
        }
    };

    loadA(0);
    loadB(0);
    int k0 = 0;
    for (;;) {
        storeA();
        storeB();
        __syncthreads();

        const int kn = k0 + 32;
        const bool more = (kn < K);
        if (more) {
            loadA(kn);                 // overlap next-tile global loads with wmma
            loadB(kn);
            if (kn + 32 < K) {         // L2 prefetch of tile k+2 (global_prefetch_b8)
                int c = tid;
                if (TA == 0) {
                    int m = c >> 3, k4 = c & 7;
                    __builtin_prefetch(A + (size_t)(m0 + m) * lda + (kn + 32 + k4 * 4), 0, 1);
                } else {
                    int k = c >> 5, m4 = c & 31;
                    __builtin_prefetch(A + (size_t)(kn + 32 + k) * lda + (m0 + m4 * 4), 0, 1);
                }
            }
        }

        v16h afr[4];
        #pragma unroll
        for (int f = 0; f < 4; ++f)
            afr[f] = frag_a(As, wr * 64 + f * 16 + lane15, kh);
        v16h bfr[2];
        #pragma unroll
        for (int g = 0; g < 2; ++g)
            bfr[g] = frag_b(Bs, wc * 32 + g * 16 + lane15, kh);
        #pragma unroll
        for (int f = 0; f < 4; ++f)
            #pragma unroll
            for (int g = 0; g < 2; ++g)
                acc[f][g] = __builtin_amdgcn_wmma_f32_16x16x32_f16(
                    false, afr[f], false, bfr[g], (short)0, acc[f][g], false, false);

        if (!more) break;
        __syncthreads();
        k0 = kn;
    }

    // ---- epilogue (regular shape: no guards) ----
    #pragma unroll
    for (int f = 0; f < 4; ++f) {
        const int rbase = m0 + wr * 64 + f * 16 + (kh << 3);
        #pragma unroll
        for (int g = 0; g < 2; ++g) {
            const int col = o0 + wc * 32 + g * 16 + lane15;
            const float bv = bias ? bias[col] : 0.0f;
            #pragma unroll
            for (int i = 0; i < 8; ++i)
                C[(size_t)(rbase + i) * ldc + col] = acc[f][g][i] + bv;
        }
    }
}

// ---------------------------------------------------------------------------
// General (guarded) GEMM for ragged shapes: K=3, K=16, M=8, O=64|16|32 ...
// Branchless clamped-address staging to avoid divergent-exec load loops.
// ---------------------------------------------------------------------------
template<int TA, int TB>
__global__ __launch_bounds__(256)
void gemm_wmma(const float* __restrict__ A, const float* __restrict__ Bm,
               const float* __restrict__ bias, float* __restrict__ C,
               int M, int K, int O, int lda, int ldb, int ldc)
{
    __shared__ f16 As[128 * LDSLD];
    __shared__ f16 Bs[128 * LDSLD];

    const int tid    = threadIdx.x;
    const int wave   = tid >> 5;
    const int lane   = tid & 31;
    const int lane15 = lane & 15;
    const int kh     = lane >> 4;
    const int wr     = wave >> 2;
    const int wc     = wave & 3;
    const int m0     = blockIdx.y * 128;
    const int o0     = blockIdx.x * 128;

    v8f acc[4][2];
    #pragma unroll
    for (int f = 0; f < 4; ++f)
        #pragma unroll
        for (int g = 0; g < 2; ++g)
            #pragma unroll
            for (int j = 0; j < 8; ++j) acc[f][g][j] = 0.0f;

    for (int k0 = 0; k0 < K; k0 += 32) {
        // stage A (branchless clamp)
        #pragma unroll
        for (int i = tid; i < 128 * 32; i += 256) {
            int m, k;
            if (TA == 0) { m = i >> 5; k = i & 31; }
            else         { m = i & 127; k = i >> 7; }       // m fastest -> coalesced
            int gm = m0 + m, gk = k0 + k;
            int cm = (gm < M) ? gm : (M - 1);
            int ck = (gk < K) ? gk : (K - 1);
            float v = (TA == 0) ? A[(size_t)cm * lda + ck] : A[(size_t)ck * lda + cm];
            As[m * LDSLD + k] = (f16)((gm < M && gk < K) ? v : 0.0f);
        }
        // stage B
        #pragma unroll
        for (int i = tid; i < 128 * 32; i += 256) {
            int o, k;
            if (TB == 0) { o = i >> 5; k = i & 31; }
            else         { o = i & 127; k = i >> 7; }
            int go = o0 + o, gk = k0 + k;
            int co = (go < O) ? go : (O - 1);
            int ck = (gk < K) ? gk : (K - 1);
            float v = (TB == 0) ? Bm[(size_t)co * ldb + ck] : Bm[(size_t)ck * ldb + co];
            Bs[o * LDSLD + k] = (f16)((go < O && gk < K) ? v : 0.0f);
        }
        __syncthreads();

        v16h afr[4];
        #pragma unroll
        for (int f = 0; f < 4; ++f)
            afr[f] = frag_a(As, wr * 64 + f * 16 + lane15, kh);
        v16h bfr[2];
        #pragma unroll
        for (int g = 0; g < 2; ++g)
            bfr[g] = frag_b(Bs, wc * 32 + g * 16 + lane15, kh);
        #pragma unroll
        for (int f = 0; f < 4; ++f)
            #pragma unroll
            for (int g = 0; g < 2; ++g)
                acc[f][g] = __builtin_amdgcn_wmma_f32_16x16x32_f16(
                    false, afr[f], false, bfr[g], (short)0, acc[f][g], false, false);
        __syncthreads();
    }

    #pragma unroll
    for (int f = 0; f < 4; ++f) {
        const int rbase = m0 + wr * 64 + f * 16 + (kh << 3);
        #pragma unroll
        for (int g = 0; g < 2; ++g) {
            int col = o0 + wc * 32 + g * 16 + lane15;
            if (col >= O) continue;
            float bv = bias ? bias[col] : 0.0f;
            #pragma unroll
            for (int i = 0; i < 8; ++i) {
                int r = rbase + i;
                if (r < M) C[(size_t)r * ldc + col] = acc[f][g][i] + bv;
            }
        }
    }
}

// ---------------------------------------------------------------------------
// BatchNorm (training mode, biased var) split into partial / finalize / apply.
// ---------------------------------------------------------------------------
__global__ void zero_two(float* a, float* b, int n) {
    int i = blockIdx.x * blockDim.x + threadIdx.x;
    if (i < n) { a[i] = 0.0f; b[i] = 0.0f; }
}

__global__ void bn_partial(const float* __restrict__ X, float* __restrict__ sum,
                           float* __restrict__ sumsq, int Mrows, int C)
{
    int c = blockIdx.x * blockDim.x + threadIdx.x;
    if (c >= C) return;
    float s = 0.0f, s2 = 0.0f;
    for (int r = blockIdx.y; r < Mrows; r += gridDim.y) {
        float v = X[(size_t)r * C + c];
        s += v; s2 += v * v;
    }
    atomicAdd(&sum[c], s);
    atomicAdd(&sumsq[c], s2);
}

__global__ void bn_finalize(const float* __restrict__ sum, const float* __restrict__ sumsq,
                            float* __restrict__ mean, float* __restrict__ rstd,
                            int Mrows, int C)
{
    int c = blockIdx.x * blockDim.x + threadIdx.x;
    if (c >= C) return;
    float m = sum[c] / (float)Mrows;
    float v = sumsq[c] / (float)Mrows - m * m;
    mean[c] = m;
    rstd[c] = rsqrtf(v + 1e-5f);
}

// act: 0=none, 1=leaky(0.2), 2=relu. Optional residual added AFTER activation.
__global__ void bn_apply(const float* __restrict__ X, const float* __restrict__ mean,
                         const float* __restrict__ rstd, const float* __restrict__ g,
                         const float* __restrict__ bb, const float* __restrict__ resid,
                         float* __restrict__ Y, size_t total, int C, int act)
{
    size_t i = (size_t)blockIdx.x * blockDim.x + threadIdx.x;
    size_t stride = (size_t)gridDim.x * blockDim.x;
    for (; i < total; i += stride) {
        int c = (int)(i % C);
        float y = g[c] * (X[i] - mean[c]) * rstd[c] + bb[c];
        if (act == 1)      y = (y > 0.0f) ? y : 0.2f * y;
        else if (act == 2) y = (y > 0.0f) ? y : 0.0f;
        if (resid) y += resid[i];
        Y[i] = y;
    }
}

// ---------------------------------------------------------------------------
// Pooling / concat / elementwise helpers
// ---------------------------------------------------------------------------
__global__ void channel_max(const float* __restrict__ X, float* __restrict__ out, int C) {
    int c = blockIdx.x * blockDim.x + threadIdx.x;
    int b = blockIdx.y;
    if (c >= C) return;
    const float* p = X + (size_t)b * NPTS * C + c;
    float m = -3.402823e38f;
    for (int n = 0; n < NPTS; ++n) m = fmaxf(m, p[(size_t)n * C]);
    out[b * C + c] = m;
}

// out[r, 0:256] = low[b, :]; out[r, 256:256+Cf] = F[r, :]
__global__ void concat_low(const float* __restrict__ low, const float* __restrict__ F,
                           float* __restrict__ out, int Cf)
{
    const int Cout = 256 + Cf;
    size_t total = (size_t)MTOT * Cout;
    size_t i = (size_t)blockIdx.x * blockDim.x + threadIdx.x;
    size_t stride = (size_t)gridDim.x * blockDim.x;
    for (; i < total; i += stride) {
        size_t r = i / Cout;
        int c = (int)(i % Cout);
        int b = (int)(r >> 11);
        out[i] = (c < 256) ? low[b * 256 + c] : F[r * Cf + (c - 256)];
    }
}

__global__ void concat2(const float* __restrict__ X0, const float* __restrict__ X1,
                        float* __restrict__ out, int C0, int C1)
{
    const int Cout = C0 + C1;
    size_t total = (size_t)MTOT * Cout;
    size_t i = (size_t)blockIdx.x * blockDim.x + threadIdx.x;
    size_t stride = (size_t)gridDim.x * blockDim.x;
    for (; i < total; i += stride) {
        size_t r = i / Cout;
        int c = (int)(i % Cout);
        out[i] = (c < C0) ? X0[r * C0 + c] : X1[r * C1 + (c - C0)];
    }
}

__global__ void ew_sub(const float* __restrict__ X, const float* __restrict__ Y,
                       float* __restrict__ Z, size_t n)
{
    size_t i = (size_t)blockIdx.x * blockDim.x + threadIdx.x;
    size_t stride = (size_t)gridDim.x * blockDim.x;
    for (; i < n; i += stride) Z[i] = X[i] - Y[i];
}

// ---------------------------------------------------------------------------
// Attention softmax (rows) + L1 column renorm, in-place on E[2048,2048]
// ---------------------------------------------------------------------------
__global__ __launch_bounds__(256) void softmax_row(float* __restrict__ E) {
    __shared__ float red[256];
    float* p = E + (size_t)blockIdx.x * NPTS;
    int t = threadIdx.x;
    float mx = -3.402823e38f;
    for (int j = t; j < NPTS; j += 256) mx = fmaxf(mx, p[j]);
    red[t] = mx; __syncthreads();
    for (int s = 128; s > 0; s >>= 1) {
        if (t < s) red[t] = fmaxf(red[t], red[t + s]);
        __syncthreads();
    }
    mx = red[0]; __syncthreads();
    float sum = 0.0f;
    for (int j = t; j < NPTS; j += 256) {
        float e = __expf(p[j] - mx);
        p[j] = e;
        sum += e;
    }
    red[t] = sum; __syncthreads();
    for (int s = 128; s > 0; s >>= 1) {
        if (t < s) red[t] += red[t + s];
        __syncthreads();
    }
    float inv = 1.0f / red[0];
    for (int j = t; j < NPTS; j += 256) p[j] *= inv;
}

__global__ void attn_colsum(const float* __restrict__ E, float* __restrict__ sums) {
    int m = blockIdx.x * blockDim.x + threadIdx.x;
    if (m >= NPTS) return;
    float s = 0.0f;
    for (int n = 0; n < NPTS; ++n) s += E[(size_t)n * NPTS + m];
    sums[m] = s;
}

__global__ void attn_renorm(float* __restrict__ E, const float* __restrict__ sums) {
    size_t total = (size_t)NPTS * NPTS;
    size_t i = (size_t)blockIdx.x * blockDim.x + threadIdx.x;
    size_t stride = (size_t)gridDim.x * blockDim.x;
    for (; i < total; i += stride) {
        int m = (int)(i & (NPTS - 1));
        E[i] = E[i] / (1e-9f + sums[m]);
    }
}

// ---------------------------------------------------------------------------
// Final [8,2560] vector assembly: [low0 | max0 | low1 | max1]
// ---------------------------------------------------------------------------
__global__ void pack_final(const float* __restrict__ low0, const float* __restrict__ max0,
                           const float* __restrict__ low1, const float* __restrict__ max1,
                           float* __restrict__ out)
{
    int i = blockIdx.x * blockDim.x + threadIdx.x;
    if (i >= NBATCH * 2560) return;
    int b = i / 2560, j = i % 2560;
    float v;
    if (j < 256)        v = low0[b * 256 + j];
    else if (j < 1280)  v = max0[b * 1024 + (j - 256)];
    else if (j < 1536)  v = low1[b * 256 + (j - 1280)];
    else                v = max1[b * 1024 + (j - 1536)];
    out[i] = v;
}

// ---------------------------------------------------------------------------
// Host orchestration
// ---------------------------------------------------------------------------
// Input index map (depth-first insertion order of setup_inputs() dict):
//  0:x  | enc: 1 c1.w 2 c1.b 3 b1.g 4 b1.b 5 c2.w 6 c2.b 7 b2.g 8 b2.b
//  9 c3.w 10 c3.b 11 b3.g 12 b3.b 13 c4.w 14 c4.b 15 b4.g 16 b4.b
//  17 c5.w 18 c5.b 19 b5.g 20 b5.b 21 c6.w 22 c6.b 23 b6.g 24 b6.b
//  enc1: 25 c1.w 26 c1.b | oa64 @27 (wqk,v.w,v.b,t.w,t.b,bn.g,bn.b)
//  34 c2.w 35 c2.b 36 c3.w 37 c3.b | ca1 @38 | ca2 @45
//  52 c4.w 53 c4.b 54 c5.w 55 c5.b | ca3 @56 | ca4 @63
//  70 c6.w 71 c6.b 72 c7.w 73 c7.b
//  74-87: b1..b7 (g,b pairs) | 88 fc.w 89 fc.b 90 bn.g 91 bn.b

extern "C" void kernel_launch(void* const* d_in, const int* in_sizes, int n_in,
                              void* d_out, int out_size, void* d_ws, size_t ws_size,
                              hipStream_t stream)
{
    (void)in_sizes; (void)n_in; (void)out_size; (void)ws_size;

    auto in = [&](int i) -> const float* { return (const float*)d_in[i]; };

    // ---- workspace carve-out (floats) ----
    float* W = (float*)d_ws;
    size_t off = 0;
    auto alloc = [&](size_t n) { float* p = W + off; off += n; return p; };
    const size_t BIG = (size_t)MTOT * 1024;
    float* big0 = alloc(BIG);
    float* big1 = alloc(BIG);
    float* big2 = alloc(BIG);
    float* big3 = alloc(BIG);
    float* qbuf = alloc((size_t)MTOT * 128);
    float* ebuf = alloc((size_t)NPTS * NPTS);
    float* sums = alloc(NPTS);
    float* ssum = alloc(2560);
    float* ssq  = alloc(2560);
    float* smean= alloc(2560);
    float* srstd= alloc(2560);
    float* lowE = alloc(NBATCH * 256);
    float* maxE = alloc(NBATCH * 1024);
    float* lowT = alloc(NBATCH * 256);
    float* maxT = alloc(NBATCH * 1024);
    float* fin  = alloc(NBATCH * 2560);
    float* fco  = alloc(NBATCH * 2560);

    auto is_regular = [](int M, int K, int O, int lda, int ldb) {
        return (M % 128 == 0) && (O % 128 == 0) && (K % 32 == 0) &&
               ((lda & 3) == 0) && ((ldb & 3) == 0);
    };
    auto gemm00 = [&](const float* A, const float* Bm, const float* bias, float* C,
                      int M, int K, int O, int lda, int ldb, int ldc) {
        dim3 g((O + 127) / 128, (M + 127) / 128);
        if (is_regular(M, K, O, lda, ldb))
            gemm_wmma_pipe<0, 0><<<g, 256, 0, stream>>>(A, Bm, bias, C, M, K, O, lda, ldb, ldc);
        else
            gemm_wmma<0, 0><<<g, 256, 0, stream>>>(A, Bm, bias, C, M, K, O, lda, ldb, ldc);
    };
    auto gemm11 = [&](const float* A, const float* Bm, const float* bias, float* C,
                      int M, int K, int O, int lda, int ldb, int ldc) {
        dim3 g((O + 127) / 128, (M + 127) / 128);
        if (is_regular(M, K, O, lda, ldb))
            gemm_wmma_pipe<1, 1><<<g, 256, 0, stream>>>(A, Bm, bias, C, M, K, O, lda, ldb, ldc);
        else
            gemm_wmma<1, 1><<<g, 256, 0, stream>>>(A, Bm, bias, C, M, K, O, lda, ldb, ldc);
    };
    auto bn_full = [&](const float* X, int Mrows, int C, const float* g, const float* bb,
                       float* Y, int act, const float* resid) {
        int cb = (C + 255) / 256;
        zero_two<<<cb, 256, 0, stream>>>(ssum, ssq, C);
        bn_partial<<<dim3(cb, 32), 256, 0, stream>>>(X, ssum, ssq, Mrows, C);
        bn_finalize<<<cb, 256, 0, stream>>>(ssum, ssq, smean, srstd, Mrows, C);
        size_t total = (size_t)Mrows * C;
        int blocks = (int)((total + 255) / 256);
        bn_apply<<<blocks, 256, 0, stream>>>(X, smean, srstd, g, bb, resid, Y, total, C, act);
    };
    auto conv_bn = [&](const float* X, float* Y, int wi, int bi, int gi, int bbi,
                       int K, int O) {
        gemm00(X, in(wi), in(bi), Y, MTOT, K, O, K, K, O);
        bn_full(Y, MTOT, O, in(gi), in(bbi), Y, /*lrelu*/1, nullptr);
    };
    // offset_attn: Out = X + relu(bn(conv_t(X - attn_apply(X)))). temps tmpV, tmpXr.
    auto oa = [&](const float* X, float* Out, float* tmpV, float* tmpXr, int base, int C) {
        int C4 = C / 4;
        gemm00(X, in(base + 0), nullptr,       qbuf, MTOT, C, C4, C, C, C4);  // q (wqk)
        gemm00(X, in(base + 1), in(base + 2),  tmpV, MTOT, C, C,  C, C, C);  // v
        for (int b = 0; b < NBATCH; ++b) {
            const float* Qb = qbuf + (size_t)b * NPTS * C4;
            const float* Vb = tmpV + (size_t)b * NPTS * C;
            float*       Rb = tmpXr + (size_t)b * NPTS * C;
            // energy = Q * Q^T   [2048,2048]
            gemm00(Qb, Qb, nullptr, ebuf, NPTS, C4, NPTS, C4, C4, NPTS);
            softmax_row<<<NPTS, 256, 0, stream>>>(ebuf);
            attn_colsum<<<NPTS / 256, 256, 0, stream>>>(ebuf, sums);
            attn_renorm<<<4096, 256, 0, stream>>>(ebuf, sums);
            // x_r[m,c] = sum_n attn[n,m] * V[n,c]  ->  A^T * B
            gemm11(ebuf, Vb, nullptr, Rb, NPTS, NPTS, C, NPTS, C, C);
        }
        size_t total = (size_t)MTOT * C;
        int blocks = (int)((total + 255) / 256);
        ew_sub<<<blocks, 256, 0, stream>>>(X, tmpXr, tmpV, total);            // diff (v dead)
        gemm00(tmpV, in(base + 3), in(base + 4), tmpXr, MTOT, C, C, C, C, C); // t-conv
        bn_full(tmpXr, MTOT, C, in(base + 5), in(base + 6), Out, /*relu*/2, X);
    };
    auto cmax = [&](const float* X, float* out, int C) {
        channel_max<<<dim3((C + 255) / 256, NBATCH), 256, 0, stream>>>(X, out, C);
    };

    const float* x = in(0);  // [16384, 3]

    // ================= enc (ConvLayer) =================
    conv_bn(x,    big0, 1, 2, 3, 4,     3,   64);
    conv_bn(big0, big1, 5, 6, 7, 8,     64,  128);
    conv_bn(big1, big0, 9, 10, 11, 12,  128, 256);
    cmax(big0, lowE, 256);
    {
        size_t total = (size_t)MTOT * 512;
        concat_low<<<(int)((total + 255) / 256), 256, 0, stream>>>(lowE, big0, big1, 256);
    }
    conv_bn(big1, big0, 13, 14, 15, 16, 512,  512);
    conv_bn(big0, big1, 17, 18, 19, 20, 512,  1024);
    conv_bn(big1, big0, 21, 22, 23, 24, 1024, 1024);
    cmax(big0, maxE, 1024);

    // ================= enc1 (ConvLayer_Transformer) =================
    conv_bn(x, big0, 25, 26, 74, 75, 3, 64);
    oa(big0, big1, big2, big3, 27, 64);
    conv_bn(big1, big2, 34, 35, 76, 77, 64, 64);
    conv_bn(big2, big0, 36, 37, 78, 79, 64, 128);
    oa(big0, big1, big2, big3, 38, 128);          // t0 = big1
    oa(big1, big2, big0, big3, 45, 128);          // t1 = big2
    {
        size_t total = (size_t)MTOT * 256;
        concat2<<<(int)((total + 255) / 256), 256, 0, stream>>>(big1, big2, big0, 128, 128);
    }
    conv_bn(big0, big1, 52, 53, 80, 81, 256, 256);  // x256 = big1
    cmax(big1, lowT, 256);
    {
        size_t total = (size_t)MTOT * 512;
        concat_low<<<(int)((total + 255) / 256), 256, 0, stream>>>(lowT, big1, big0, 256);
    }
    conv_bn(big0, big1, 54, 55, 82, 83, 512, 512);  // x512 = big1
    oa(big1, big0, big2, big3, 56, 512);            // t0 = big0
    oa(big0, big2, big1, big3, 63, 512);            // t1 = big2
    {
        size_t total = (size_t)MTOT * 1024;
        concat2<<<(int)((total + 255) / 256), 256, 0, stream>>>(big0, big2, big1, 512, 512);
    }
    conv_bn(big1, big0, 70, 71, 84, 85, 1024, 1024);
    conv_bn(big0, big1, 72, 73, 86, 87, 1024, 1024);
    cmax(big1, maxT, 1024);

    // ================= head =================
    pack_final<<<(NBATCH * 2560 + 255) / 256, 256, 0, stream>>>(lowE, maxE, lowT, maxT, fin);
    gemm00(fin, in(88), in(89), fco, NBATCH, 2560, 2560, 2560, 2560, 2560);
    bn_full(fco, NBATCH, 2560, in(90), in(91), (float*)d_out, /*relu*/2, nullptr);
}
